// FF_BICONV3_58025008169049
// MI455X (gfx1250) — 8-bit matrix operand order repaired and checked by reading, compiled, not run on hardware
//
#include <hip/hip_runtime.h>
#include <cstdint>


typedef int v8i __attribute__((ext_vector_type(8)));

#define NN   16
#define CC   128
#define HH   128
#define WW   128
#define NHW  (NN * HH * WW)
#define TILE_W 66
#define TILE_BYTES (3 * TILE_W * CC)

__device__ __forceinline__ void async_load_b128_to_lds(void* lds_ptr,
                                                       const void* gptr) {
  unsigned lds_off = (unsigned)(size_t)lds_ptr;
  unsigned long long ga = (unsigned long long)(size_t)gptr;
  asm volatile("global_load_async_to_lds_b128 %0, %1, off"
               :
               : "v"(lds_off), "v"(ga)
               : "memory");
}
__device__ __forceinline__ void wait_asynccnt0() {
  asm volatile("s_wait_asynccnt 0x0" ::: "memory");
}

__global__ __launch_bounds__(256) void FF_bpack_x(const float* __restrict__ x,
                                                  int8_t* __restrict__ xb) {
  unsigned tid = blockIdx.x * 256u + threadIdx.x;
  unsigned w  = tid & 127u;
  unsigned c4 = (tid >> 7) & 31u;
  unsigned h  = (tid >> 12) & 127u;
  unsigned n  = tid >> 19;
  unsigned c0 = c4 * 4u;
  signed char s[4];
#pragma unroll
  for (int j = 0; j < 4; ++j) {
    float v = x[(((size_t)(n * CC + c0 + j) * HH + h) * WW + w)];
    s[j] = (signed char)((v > 0.f) - (v < 0.f));
  }
  char4 pk = make_char4(s[0], s[1], s[2], s[3]);
  *(char4*)(xb + (((size_t)(n * HH + h) * WW + w) * CC + c0)) = pk;
}

__global__ __launch_bounds__(256) void FF_bpack_w(const float* __restrict__ wt,
                                                  int* __restrict__ wfrag) {
  int tap_kc = blockIdx.x;
  int tap = tap_kc >> 1, kc = tap_kc & 1;
  int kh = tap / 3, kw = tap % 3;
  int ot = threadIdx.x >> 5;
  int lane = threadIdx.x & 31;
  int khalf = (lane >> 4) * 32;
  int cout = ot * 16 + (lane & 15);
#pragma unroll
  for (int d = 0; d < 8; ++d) {
    unsigned dw = 0;
#pragma unroll
    for (int b = 0; b < 4; ++b) {
      int cin = kc * 64 + khalf + 4 * d + b;
      float v = wt[((size_t)(cout * CC + cin) * 3 + kh) * 3 + kw];
      signed char s = (signed char)((v > 0.f) - (v < 0.f));
      dw |= (unsigned)(unsigned char)s << (8 * b);
    }
    wfrag[((size_t)tap_kc * 8 + ot) * 256 + lane * 8 + d] = (int)dw;
  }
}

__global__ __launch_bounds__(256) void FF_biconv_wmma(
    const int8_t* __restrict__ xb, const int* __restrict__ wfrag,
    float* __restrict__ conv, int* __restrict__ gsum,
    unsigned long long* __restrict__ gsq) {
  __shared__ __align__(16) char tile[TILE_BYTES];
  __shared__ int lsum[CC];
  __shared__ unsigned long long lsq[CC];

  unsigned blk = blockIdx.x;
  int w0 = (int)(blk & 1u) * 64;
  int h  = (int)(blk >> 1) & 127;
  int n  = (int)(blk >> 8);
  int tid = threadIdx.x;

  if (tid < CC) { lsum[tid] = 0; lsq[tid] = 0ull; }

  int wv = tid >> 5, lane = tid & 31;

  __builtin_prefetch(wfrag + (size_t)wv * 256 + lane * 8, 0, 0);

  for (int ch = tid; ch < 198 * 8; ch += 256) {
    int r = ch >> 3, j = ch & 7;
    int kh = r / TILE_W, wi = r % TILE_W;
    int hh = h + kh - 1, ww = w0 + wi - 1;
    void* dst = tile + r * CC + j * 16;
    if (hh >= 0 && hh < HH && ww >= 0 && ww < WW) {
      async_load_b128_to_lds(
          dst, xb + ((size_t)(n * HH + hh) * WW + ww) * CC + j * 16);
    } else {
      *(int4*)dst = make_int4(0, 0, 0, 0);
    }
  }
  wait_asynccnt0();
  __syncthreads();

  int laneN = lane & 15;
  int hiOff = (lane >> 4) << 5;

  v8i acc[4] = {};

#pragma unroll
  for (int kh = 0; kh < 3; ++kh)
#pragma unroll
    for (int kw = 0; kw < 3; ++kw)
#pragma unroll
      for (int kc = 0; kc < 2; ++kc) {
        size_t fb = (((size_t)((kh * 3 + kw) * 2 + kc) * 8) + wv) * 256 + lane * 8;
        union { v8i v; int4 q[2]; } B;
        B.q[0] = *(const int4*)(wfrag + fb);
        B.q[1] = *(const int4*)(wfrag + fb + 4);
#pragma unroll
        for (int ms = 0; ms < 4; ++ms) {
          int row = kh * TILE_W + ms * 16 + laneN + kw;
          const unsigned long long* ap =
              (const unsigned long long*)(tile + row * CC + kc * 64 + hiOff);
          union { v8i v; unsigned long long u[4]; } A;
          A.u[0] = ap[0]; A.u[1] = ap[1]; A.u[2] = ap[2]; A.u[3] = ap[3];
          acc[ms] = __builtin_amdgcn_wmma_i32_16x16x64_iu8(
               true, A.v,  true, B.v, acc[ms],
               false,  false);
        }
      }

  int cout = wv * 16 + laneN;
  int mb = (lane >> 4) * 8;
  int ssum = 0;
  unsigned long long sq = 0;
  size_t pixbase = (size_t)(n * HH + h) * WW;
#pragma unroll
  for (int ms = 0; ms < 4; ++ms)
#pragma unroll
    for (int i = 0; i < 8; ++i) {
      int d = acc[ms][i];
      conv[(pixbase + (unsigned)(w0 + ms * 16 + mb + i)) * CC + cout] = (float)d;
      ssum += d;
      sq += (unsigned long long)((long long)d * (long long)d);
    }
  atomicAdd(&lsum[cout], ssum);
  atomicAdd(&lsq[cout], sq);
  __syncthreads();
  if (tid < CC) {
    atomicAdd(&gsum[tid], lsum[tid]);
    atomicAdd(&gsq[tid], lsq[tid]);
  }
}

__global__ void FF_bn_stats(const int* __restrict__ gsum,
                            const unsigned long long* __restrict__ gsq,
                            const float* __restrict__ gamma,
                            const float* __restrict__ beta,
                            float* __restrict__ scale, float* __restrict__ bias) {
  int c = threadIdx.x;
  const double cnt = (double)NHW;
  double mean = (double)gsum[c] / cnt;
  double var = (double)gsq[c] / cnt - mean * mean;
  double inv = 1.0 / sqrt(var + 1e-5);
  double sc = inv * (double)gamma[c];
  scale[c] = (float)sc;
  bias[c] = (float)((double)beta[c] - mean * sc);
}

__global__ __launch_bounds__(256) void FF_bn_finalize(
    const float* __restrict__ conv, const float* __restrict__ x,
    const float* __restrict__ scale, const float* __restrict__ bias,
    float* __restrict__ out) {
  __shared__ float t[CC * 33];
  __shared__ float ls[CC], lb[CC];
  int w0 = blockIdx.x * 32;
  int h = blockIdx.y;
  int n = blockIdx.z;
  int tid = threadIdx.x;
  if (tid < CC) { ls[tid] = scale[tid]; lb[tid] = bias[tid]; }
  size_t base = ((size_t)(n * HH + h) * WW + w0) * CC;
  for (int e = tid; e < 4096; e += 256) {
    int c = e & 127, p = e >> 7;
    t[c * 33 + p] = conv[base + e];
  }
  __syncthreads();
  for (int e = tid; e < 4096; e += 256) {
    int c = e >> 5, p = e & 31;
    float y = t[c * 33 + p] * ls[c] + lb[c];
    y = fminf(1.f, fmaxf(-1.f, y));
    size_t idx = (((size_t)(n * CC + c) * HH + h) * WW) + w0 + p;
    out[idx] = y + x[idx];
  }
}

extern "C" void kernel_launch(void* const* d_in, const int* in_sizes, int n_in,
                              void* d_out, int out_size, void* d_ws, size_t ws_size,
                              hipStream_t stream) {
  const float* x     = (const float*)d_in[0];
  const float* wt    = (const float*)d_in[1];
  const float* gamma = (const float*)d_in[2];
  const float* beta  = (const float*)d_in[3];
  float* out = (float*)d_out;

  char* ws = (char*)d_ws;
  float* conv = (float*)ws;
  int8_t* xb = (int8_t*)(ws + 134217728ull);
  int* wfrag = (int*)(ws + 167772160ull);
  int* gsum = (int*)(ws + 167919616ull);
  unsigned long long* gsq = (unsigned long long*)(ws + 167920128ull);
  float* scale = (float*)(ws + 167921152ull);
  float* bias  = (float*)(ws + 167921664ull);

  hipMemsetAsync(gsum, 0, 512 + 1024, stream);

  FF_bpack_x<<<32768, 256, 0, stream>>>(x, xb);
  FF_bpack_w<<<18, 256, 0, stream>>>(wt, wfrag);
  FF_biconv_wmma<<<4096, 256, 0, stream>>>(xb, wfrag, conv, gsum, gsq);
  FF_bn_stats<<<1, 128, 0, stream>>>(gsum, gsq, gamma, beta, scale, bias);
  FF_bn_finalize<<<dim3(4, 128, 16), 256, 0, stream>>>(conv, x, scale, bias, out);
}
